// TemporalGRU_9921374453891
// MI455X (gfx1250) — compile-verified, matrix operand order checked by reading, not run on hardware
//
#include <hip/hip_runtime.h>
#include <stdint.h>

// CDNA5 / gfx1250: wave32, WMMA 16x16x32 bf16 (f32 accum).
typedef __attribute__((ext_vector_type(16))) __bf16 v16bf;
typedef __attribute__((ext_vector_type(8)))  float  v8f;

union ABReg { v16bf v; unsigned u[8]; };

__device__ __forceinline__ unsigned pk_bf16(float a, float b) {
  unsigned ua = __float_as_uint(a), ub = __float_as_uint(b);
  ua = (ua + 0x7FFFu + ((ua >> 16) & 1u)) >> 16;          // RNE truncate
  ub = (ub + 0x7FFFu + ((ub >> 16) & 1u)) >> 16;
  return (ua & 0xFFFFu) | (ub << 16);
}
__device__ __forceinline__ unsigned short cvt_bf16(float a) {
  unsigned ua = __float_as_uint(a);
  ua = (ua + 0x7FFFu + ((ua >> 16) & 1u)) >> 16;
  return (unsigned short)ua;
}

// ---------------------------------------------------------------------------
// Phase 0: one-shot fp32 -> packed bf16 (RNE). Bandwidth-trivial.
// ---------------------------------------------------------------------------
__global__ void __launch_bounds__(256)
pack_bf16_kernel(const float* __restrict__ in, unsigned* __restrict__ out,
                 long npairs) {
  long i = (long)blockIdx.x * blockDim.x + threadIdx.x;
  if (i < npairs) out[i] = pk_bf16(in[2 * i], in[2 * i + 1]);
}

// Load one 16(M/N)x32(K) bf16 operand tile from a row-major bf16 matrix
// (ld = 128 elements). Striping per ISA 7.12.2 (16-bit 16x32):
//   lanes 0-15 : row = row_base + lane,     K = {c..c+7, c+16..c+23}, c = 32*kt
//   lanes 16-31: row = row_base + lane-16,  K = {c+8..c+15, c+24..c+31}
// Byte offset (row*128 + c0)*2 is 16B-aligned -> two b128 loads, no VALU.
__device__ __forceinline__ ABReg load_tile_bf16(const unsigned short* __restrict__ mat,
                                                int row_base, int kt, int lane) {
  ABReg r;
  const int rr = row_base + (lane & 15);
  const int c0 = kt * 32 + ((lane & 16) ? 8 : 0);
  const uint4* q = (const uint4*)(mat + rr * 128 + c0);
  uint4 lo = q[0];     // K = c0 .. c0+7
  uint4 hi = q[2];     // K = c0+16 .. c0+23  (+32 bytes)
  r.u[0] = lo.x; r.u[1] = lo.y; r.u[2] = lo.z; r.u[3] = lo.w;
  r.u[4] = hi.x; r.u[5] = hi.y; r.u[6] = hi.z; r.u[7] = hi.w;
  return r;
}

// ---------------------------------------------------------------------------
// Phase 1: x_gates[N,384] = A[N,128] @ W_ih^T[128,384] + b_ih   (WMMA bf16)
// One block per 16-row strip; 8 waves, 3 column tiles (of 16) each.
// ---------------------------------------------------------------------------
__global__ void __launch_bounds__(256)
xgates_gemm(const unsigned short* __restrict__ A,    // bf16 [N,128]
            const unsigned short* __restrict__ Wih,  // bf16 [384,128]
            const float* __restrict__ bih, float* __restrict__ XG, int N) {
  const int lane = threadIdx.x & 31;
  const int wave = threadIdx.x >> 5;          // 0..7
  const int row0 = blockIdx.x * 16;
  if (row0 >= N) return;

  v8f acc[3];
#pragma unroll
  for (int c = 0; c < 3; ++c) {
    const float bv = bih[(wave * 3 + c) * 16 + (lane & 15)];
#pragma unroll
    for (int e = 0; e < 8; ++e) acc[c][e] = bv;   // bias broadcast to all rows
  }

#pragma unroll
  for (int kt = 0; kt < 4; ++kt) {                // K = 128 in 4 chunks of 32
    ABReg a = load_tile_bf16(A, row0, kt, lane);
#pragma unroll
    for (int c = 0; c < 3; ++c) {
      ABReg b = load_tile_bf16(Wih, (wave * 3 + c) * 16, kt, lane); // B = W^T
      acc[c] = __builtin_amdgcn_wmma_f32_16x16x32_bf16(
          false, a.v, false, b.v, (short)0, acc[c], false, false);
    }
  }

  // D layout: element e, lanes 0-15 -> M=e; lanes 16-31 -> M=e+8; N = lane%16
  const int mbase = (lane & 16) ? 8 : 0;
#pragma unroll
  for (int c = 0; c < 3; ++c) {
    const int n = (wave * 3 + c) * 16 + (lane & 15);
#pragma unroll
    for (int e = 0; e < 8; ++e)
      XG[(size_t)(row0 + mbase + e) * 384 + n] = acc[c][e];
  }
}

// ---------------------------------------------------------------------------
// Phase 2: sequential GRU scan. One workgroup, 384 threads (12 waves).
// W_hh lives in registers as WMMA B-tiles (no weight traffic in the loop).
// Per step the 128->384 matvec is 8 WMMA issues/wave with a one-live-row A.
// ---------------------------------------------------------------------------
__global__ void __launch_bounds__(384)
gru_scan(const float* __restrict__ XG,
         const unsigned short* __restrict__ Whh,    // bf16 [384,128]
         const float* __restrict__ bhh, const float* __restrict__ Art,
         float* __restrict__ Out, int N) {
  const int tid  = threadIdx.x;
  const int lane = tid & 31;
  const int wave = tid >> 5;                    // 0..11, owns N-tiles 2w, 2w+1

  __shared__ unsigned Hpk[64];                  // h as 128 packed bf16
  __shared__ float    hg[384];
  __shared__ float    xg[384];

  // Resident B-tiles of W_hh^T: 2 N-tiles x 4 K-chunks per wave.
  ABReg B[2][4];
#pragma unroll
  for (int c = 0; c < 2; ++c)
#pragma unroll
    for (int kt = 0; kt < 4; ++kt)
      B[c][kt] = load_tile_bf16(Whh, (2 * wave + c) * 16, kt, lane);

  // b_hh folded into accumulator row 0 (element 0, lanes 0-15).
  float bias0[2];
#pragma unroll
  for (int c = 0; c < 2; ++c)
    bias0[c] = (lane < 16) ? bhh[(2 * wave + c) * 16 + lane] : 0.0f;

  float h = 0.0f;                               // lanes tid<128 each own h_j
  if (tid < 64) Hpk[tid] = 0u;
  __syncthreads();

  for (int t = 0; t < N; ++t) {
    const float x = XG[(size_t)t * 384 + tid];
    xg[tid] = x;
    if (t + 8 < N)                              // global_prefetch_b8
      __builtin_prefetch(&XG[(size_t)(t + 8) * 384 + tid], 0, 0);

    // hg = h @ W_hh^T + b_hh via WMMA; only row 0 of each D tile is live.
    v8f acc[2];
#pragma unroll
    for (int c = 0; c < 2; ++c) {
#pragma unroll
      for (int e = 0; e < 8; ++e) acc[c][e] = 0.0f;
      acc[c][0] = bias0[c];
    }
#pragma unroll
    for (int kt = 0; kt < 4; ++kt) {
      // A tile: row 0 = h[32kt..32kt+31] (bf16 from LDS), rows 1-15 zero.
      ABReg a;
      const int  base  = kt * 16 + ((lane & 16) ? 4 : 0);
      const bool valid = ((lane & 15) == 0);    // only M=0 rows carry data
#pragma unroll
      for (int i = 0; i < 4; ++i) a.u[i]     = valid ? Hpk[base + i]     : 0u;
#pragma unroll
      for (int i = 0; i < 4; ++i) a.u[4 + i] = valid ? Hpk[base + 8 + i] : 0u;
#pragma unroll
      for (int c = 0; c < 2; ++c)
        acc[c] = __builtin_amdgcn_wmma_f32_16x16x32_bf16(
            false, a.v, false, B[c][kt].v, (short)0, acc[c], false, false);
    }
    if (lane < 16) {                            // D row 0: element 0, N = lane
#pragma unroll
      for (int c = 0; c < 2; ++c) hg[(2 * wave + c) * 16 + lane] = acc[c][0];
    }
    __syncthreads();                            // hg (and xg) visible

    if (tid < 128) {
      const float r = 1.0f / (1.0f + __expf(-(xg[tid]       + hg[tid])));
      const float z = 1.0f / (1.0f + __expf(-(xg[128 + tid] + hg[128 + tid])));
      const float n = tanhf(xg[256 + tid] + r * hg[256 + tid]);
      h = (1.0f - z) * n + z * h;
      ((unsigned short*)Hpk)[tid] = cvt_bf16(h);
      Out[(size_t)t * 128 + tid] = Art[(size_t)t * 128 + tid] + h;  // residual
    }
    __syncthreads();                            // Hpk/xg WAR for next step
  }
}

// ---------------------------------------------------------------------------
extern "C" void kernel_launch(void* const* d_in, const int* in_sizes, int n_in,
                              void* d_out, int out_size, void* d_ws, size_t ws_size,
                              hipStream_t stream) {
  const float* art = (const float*)d_in[0];   // [N,128]
  const float* wih = (const float*)d_in[1];   // [384,128]
  const float* whh = (const float*)d_in[2];   // [384,128]
  const float* bih = (const float*)d_in[3];   // [384]
  const float* bhh = (const float*)d_in[4];   // [384]
  float* out = (float*)d_out;                 // [N,128]

  const int N = in_sizes[0] / 128;

  // Workspace layout: XG fp32 [N,384] | A bf16 [N,128] | Wih bf16 | Whh bf16
  float* XG = (float*)d_ws;
  unsigned short* Abf  = (unsigned short*)((char*)d_ws + (size_t)N * 384 * 4);
  unsigned short* Wihb = Abf + (size_t)N * 128;
  unsigned short* Whhb = Wihb + 384 * 128;

  // Phase 0: pre-pack fp32 -> bf16 (one elementwise pass each).
  {
    long np_a = (long)N * 64;                 // pairs
    pack_bf16_kernel<<<(unsigned)((np_a + 255) / 256), 256, 0, stream>>>(
        art, (unsigned*)Abf, np_a);
    pack_bf16_kernel<<<(384 * 64 + 255) / 256, 256, 0, stream>>>(
        wih, (unsigned*)Wihb, 384 * 64);
    pack_bf16_kernel<<<(384 * 64 + 255) / 256, 256, 0, stream>>>(
        whh, (unsigned*)Whhb, 384 * 64);
  }

  // Phase 1: big WMMA GEMM for input gates.
  xgates_gemm<<<(N + 15) / 16, 256, 0, stream>>>(Abf, Wihb, bih, XG, N);

  // Phase 2: latency-bound sequential scan on one WGP.
  gru_scan<<<1, 384, 0, stream>>>(XG, Whhb, bhh, art, out, N);
}
